// LinearAttention_88467736363626
// MI455X (gfx1250) — matrix operand order repaired and checked by reading, not compiled, not run on hardware
//
#include <hip/hip_runtime.h>
#include <hip/hip_bf16.h>

// Linear attention, MI455X (gfx1250), wave32 WMMA bf16 path.
// Pass 1: KV[n,h,64,64] += phi(K)^T @ (V/S), Ksum[n,h,64] += col-sums of phi(K)
// Pass 2: out = (phi(Q) @ KV) * S / (phi(Q).Ksum + eps)
// Bandwidth-bound (268 MB @ 23.3 TB/s ~= 11.5 us floor): all global streams are
// coalesced 256B-per-16-lane accesses with non-temporal hints; KV/Ksum stay L2.

typedef __attribute__((ext_vector_type(16))) __bf16 v16bf;
typedef __attribute__((ext_vector_type(8)))  __bf16 v8bf;
typedef __attribute__((ext_vector_type(8)))  float  v8f;
typedef __attribute__((ext_vector_type(4)))  float  v4f;

#define LA_N 4
#define LA_H 8
#define LA_D 64
#define LA_S 8192
#define LA_L 8192
#define LA_EPS 1e-6f
#define K1_ITERS 4   // S-chunks of 128 rows per block in pass 1

__device__ __forceinline__ float la_feat(float x) {
  // elu(x) + 1
  return x > 0.0f ? (x + 1.0f) : __expf(x);
}

// Build a 16-half WMMA operand from two 16B-aligned LDS segments.
__device__ __forceinline__ v16bf la_frag(const __bf16* p0, const __bf16* p1) {
  v8bf lo = *(const v8bf*)p0;
  v8bf hi = *(const v8bf*)p1;
  v16bf r;
#pragma unroll
  for (int i = 0; i < 8; ++i) { r[i] = lo[i]; r[i + 8] = hi[i]; }
  return r;
}

// ---------------------------------------------------------------------------
// Workspace zero (harness poisons d_ws with 0xAA).
// ---------------------------------------------------------------------------
__global__ void la_zero_kernel(float* __restrict__ p, int n) {
  int i = blockIdx.x * blockDim.x + threadIdx.x;
  if (i < n) p[i] = 0.0f;
}

// ---------------------------------------------------------------------------
// Pass 1: per (n,h), accumulate KV (64x64, f32) and Ksum (64, f32).
// grid = (S/(128*K1_ITERS), N*H), block = 128 threads (4 waves).
// ---------------------------------------------------------------------------
__global__ void __launch_bounds__(128)
la_kv_kernel(const float* __restrict__ Kin, const float* __restrict__ Vin,
             float* __restrict__ KV, float* __restrict__ Ksum) {
  const int nh = blockIdx.y;           // n*H + h
  const int n  = nh >> 3;
  const int h  = nh & 7;
  const int t  = threadIdx.x;

  constexpr int STR = 136;             // bf16 stride: 16B aligned, bank-staggered
  __shared__ __align__(16) __bf16 sKT[64 * STR];   // [d][s_local]  phi(K)^T
  __shared__ __align__(16) __bf16 sVT[64 * STR];   // [e][s_local]  (V/S)^T

  const int wave = t >> 5, lane = t & 31;
  const int lh = lane >> 4, lm = lane & 15;
  const int drow  = wave * 16 + lm;
  const int rlane = t >> 4;            // 0..7 : row group for coalesced loads
  const int c4    = t & 15;            // float4 column within a 256B row

  const float inv_len = 1.0f / (float)LA_S;
  v8f acc0 = {}, acc1 = {}, acc2 = {}, acc3 = {};
  float ksumAcc = 0.0f;

  for (int ci = 0; ci < K1_ITERS; ++ci) {
    const int s0 = (blockIdx.x * K1_ITERS + ci) * 128;
    if (ci) __syncthreads();           // protect LDS reuse across chunks

    // Coalesced stream: 16 lanes cover one contiguous 256B row; 8 rows/iter.
#pragma unroll
    for (int i = 0; i < 16; ++i) {
      const int sl = i * 8 + rlane;    // local s in [0,128)
      const size_t base =
          (((size_t)n * LA_S + s0 + sl) * LA_H + h) * LA_D + c4 * 4;
      v4f kk = __builtin_nontemporal_load((const v4f*)(Kin + base));
      v4f vv = __builtin_nontemporal_load((const v4f*)(Vin + base));
      const int d = c4 * 4;
      sKT[(d + 0) * STR + sl] = (__bf16)la_feat(kk.x);
      sKT[(d + 1) * STR + sl] = (__bf16)la_feat(kk.y);
      sKT[(d + 2) * STR + sl] = (__bf16)la_feat(kk.z);
      sKT[(d + 3) * STR + sl] = (__bf16)la_feat(kk.w);
      sVT[(d + 0) * STR + sl] = (__bf16)(vv.x * inv_len);
      sVT[(d + 1) * STR + sl] = (__bf16)(vv.y * inv_len);
      sVT[(d + 2) * STR + sl] = (__bf16)(vv.z * inv_len);
      sVT[(d + 3) * STR + sl] = (__bf16)(vv.w * inv_len);
    }
    __syncthreads();

    // Ksum partial: one phi(K)^T row per thread (waves 0-1, wave-uniform).
    if (t < 64) {
#pragma unroll 8
      for (int j = 0; j < 128; ++j) ksumAcc += (float)sKT[t * STR + j];
    }

    // WMMA: wave w owns d-tile w; 4 k-steps of 32 over s; 4 e-tiles.
    // All 5 fragments loaded into distinct values before any WMMA issues,
    // so the ds_load_b128s clause together and WMMAs drain on partial waits.
#pragma unroll
    for (int kd = 0; kd < 4; ++kd) {
      const int sa = kd * 32 + lh * 16;
      const int sb = kd * 32 + lh * 16;
      v16bf a  = la_frag(&sKT[drow * STR + sa], &sKT[drow * STR + sa + 8]);
      v16bf b0 = la_frag(&sVT[(0 * 16 + lm) * STR + sb],
                         &sVT[(0 * 16 + lm) * STR + sb + 8]);
      v16bf b1 = la_frag(&sVT[(1 * 16 + lm) * STR + sb],
                         &sVT[(1 * 16 + lm) * STR + sb + 8]);
      v16bf b2 = la_frag(&sVT[(2 * 16 + lm) * STR + sb],
                         &sVT[(2 * 16 + lm) * STR + sb + 8]);
      v16bf b3 = la_frag(&sVT[(3 * 16 + lm) * STR + sb],
                         &sVT[(3 * 16 + lm) * STR + sb + 8]);
      acc0 = __builtin_amdgcn_wmma_f32_16x16x32_bf16(false, a, false, b0,
                                                     (short)0, acc0, false, false);
      acc1 = __builtin_amdgcn_wmma_f32_16x16x32_bf16(false, a, false, b1,
                                                     (short)0, acc1, false, false);
      acc2 = __builtin_amdgcn_wmma_f32_16x16x32_bf16(false, a, false, b2,
                                                     (short)0, acc2, false, false);
      acc3 = __builtin_amdgcn_wmma_f32_16x16x32_bf16(false, a, false, b3,
                                                     (short)0, acc3, false, false);
    }
  }

  if (t < 64) atomicAdd(&Ksum[nh * 64 + t], ksumAcc);

  // Accumulate partial 64x64 into global f32 KV (L2-resident, 512KB total).
  float* kvb = KV + (size_t)nh * 64 * 64;
  const int dbase = wave * 16 + lh * 8;           // C layout: lanes 16-31 -> M+8
#pragma unroll
  for (int r = 0; r < 8; ++r) {
    const int d = dbase + r;
    atomicAdd(&kvb[d * 64 + 0 * 16 + lm], acc0[r]);
    atomicAdd(&kvb[d * 64 + 1 * 16 + lm], acc1[r]);
    atomicAdd(&kvb[d * 64 + 2 * 16 + lm], acc2[r]);
    atomicAdd(&kvb[d * 64 + 3 * 16 + lm], acc3[r]);
  }
}

// ---------------------------------------------------------------------------
// Pass 2: out[l,e] = (phi(Q) @ KV) * S / (phi(Q).Ksum + eps).
// grid = (L/64, N*H), block = 128 threads (4 waves, 16 Q-rows each).
// ---------------------------------------------------------------------------
__global__ void __launch_bounds__(128)
la_out_kernel(const float* __restrict__ Qin, const float* __restrict__ KV,
              const float* __restrict__ Ksum, float* __restrict__ Out) {
  const int nh = blockIdx.y;
  const int n  = nh >> 3;
  const int h  = nh & 7;
  const int l0 = blockIdx.x * 64;
  const int t  = threadIdx.x;

  constexpr int STR  = 72;             // bf16 stride (16B aligned, staggered)
  constexpr int OSTR = 68;             // f32 stride for staged output
  __shared__ __align__(16) __bf16 sKVT[64 * STR];  // [e][d] = KV[d][e]
  __shared__ __align__(16) __bf16 sQ[64 * STR];    // [row][d] = phi(Q)
  __shared__ __align__(16) float  sOut[64 * OSTR]; // staged scaled output
  __shared__ float sKsum[64];
  __shared__ float sDen[64];

  const int rlane = t >> 4;
  const int c4    = t & 15;

  // Cache KV^T as bf16 (global side fully coalesced) + Ksum f32.
  const float* kvp = KV + (size_t)nh * 4096;
#pragma unroll
  for (int i = 0; i < 32; ++i) {
    const int flat = t + 128 * i;                  // flat = d*64 + e
    sKVT[(flat & 63) * STR + (flat >> 6)] = (__bf16)kvp[flat];
  }
  if (t < 64) {
    sKsum[t] = Ksum[nh * 64 + t];
    sDen[t] = 0.0f;
  }

  // Stream 64 Q rows, coalesced 256B-per-16-lane, fused feature map.
#pragma unroll
  for (int i = 0; i < 8; ++i) {
    const int r = i * 8 + rlane;
    const size_t base =
        (((size_t)n * LA_L + l0 + r) * LA_H + h) * LA_D + c4 * 4;
    v4f q = __builtin_nontemporal_load((const v4f*)(Qin + base));
    const int c = c4 * 4;
    sQ[r * STR + c + 0] = (__bf16)la_feat(q.x);
    sQ[r * STR + c + 1] = (__bf16)la_feat(q.y);
    sQ[r * STR + c + 2] = (__bf16)la_feat(q.z);
    sQ[r * STR + c + 3] = (__bf16)la_feat(q.w);
  }
  __syncthreads();

  // Z denominator from staged phi(Q): 2 threads per row, one ds-atomic each.
  {
    const int r  = t >> 1;
    const int c0 = (t & 1) * 32;
    float part = 0.0f;
#pragma unroll
    for (int j = 0; j < 32; ++j)
      part += (float)sQ[r * STR + c0 + j] * sKsum[c0 + j];
    atomicAdd(&sDen[r], part);                     // ds_add_f32
  }
  __syncthreads();

  // WMMA: wave w owns rows [w*16, w*16+16); 2 k-steps of 32 over d.
  const int wave = t >> 5, lane = t & 31;
  const int lh = lane >> 4, lm = lane & 15;
  const int qrow = wave * 16 + lm;

  v8f acc0 = {}, acc1 = {}, acc2 = {}, acc3 = {};
#pragma unroll
  for (int kd = 0; kd < 2; ++kd) {
    const int ca = kd * 32 + lh * 16;
    const int cb = kd * 32 + lh * 16;
    v16bf a  = la_frag(&sQ[qrow * STR + ca], &sQ[qrow * STR + ca + 8]);
    v16bf b0 = la_frag(&sKVT[(0 * 16 + lm) * STR + cb],
                       &sKVT[(0 * 16 + lm) * STR + cb + 8]);
    v16bf b1 = la_frag(&sKVT[(1 * 16 + lm) * STR + cb],
                       &sKVT[(1 * 16 + lm) * STR + cb + 8]);
    v16bf b2 = la_frag(&sKVT[(2 * 16 + lm) * STR + cb],
                       &sKVT[(2 * 16 + lm) * STR + cb + 8]);
    v16bf b3 = la_frag(&sKVT[(3 * 16 + lm) * STR + cb],
                       &sKVT[(3 * 16 + lm) * STR + cb + 8]);
    acc0 = __builtin_amdgcn_wmma_f32_16x16x32_bf16(false, a, false, b0,
                                                   (short)0, acc0, false, false);
    acc1 = __builtin_amdgcn_wmma_f32_16x16x32_bf16(false, a, false, b1,
                                                   (short)0, acc1, false, false);
    acc2 = __builtin_amdgcn_wmma_f32_16x16x32_bf16(false, a, false, b2,
                                                   (short)0, acc2, false, false);
    acc3 = __builtin_amdgcn_wmma_f32_16x16x32_bf16(false, a, false, b3,
                                                   (short)0, acc3, false, false);
  }

  // Epilogue: scale by S / (den + eps), stage in LDS, then coalesced NT store.
#pragma unroll
  for (int r = 0; r < 8; ++r) {
    const int row = wave * 16 + lh * 8 + r;
    const float z = (float)LA_S / (sDen[row] + LA_EPS);
    sOut[row * OSTR + 0 * 16 + lm] = acc0[r] * z;
    sOut[row * OSTR + 1 * 16 + lm] = acc1[r] * z;
    sOut[row * OSTR + 2 * 16 + lm] = acc2[r] * z;
    sOut[row * OSTR + 3 * 16 + lm] = acc3[r] * z;
  }
  __syncthreads();

#pragma unroll
  for (int i = 0; i < 8; ++i) {
    const int r = i * 8 + rlane;
    v4f val = *(const v4f*)&sOut[r * OSTR + c4 * 4];
    const size_t base =
        (((size_t)n * LA_L + l0 + r) * LA_H + h) * LA_D + c4 * 4;
    __builtin_nontemporal_store(val, (v4f*)(Out + base));
  }
}

// ---------------------------------------------------------------------------
extern "C" void kernel_launch(void* const* d_in, const int* in_sizes, int n_in,
                              void* d_out, int out_size, void* d_ws, size_t ws_size,
                              hipStream_t stream) {
  const float* Qp = (const float*)d_in[0];
  const float* Kp = (const float*)d_in[1];
  const float* Vp = (const float*)d_in[2];
  float* Out = (float*)d_out;

  // Workspace: KV [N*H*64*64] f32, then Ksum [N*H*64] f32 (~532 KB).
  float* KV = (float*)d_ws;
  float* Ks = KV + (size_t)LA_N * LA_H * 64 * 64;
  const int nz = LA_N * LA_H * 64 * 64 + LA_N * LA_H * 64;

  la_zero_kernel<<<(nz + 255) / 256, 256, 0, stream>>>(KV, nz);

  dim3 g1(LA_S / (128 * K1_ITERS), LA_N * LA_H);
  la_kv_kernel<<<g1, 128, 0, stream>>>(Kp, Vp, KV, Ks);

  dim3 g2(LA_L / 64, LA_N * LA_H);
  la_out_kernel<<<g2, 128, 0, stream>>>(Qp, KV, Ks, Out);
}
